// BenesBlock_10857677325066
// MI455X (gfx1250) — matrix operand order repaired and checked by reading, not compiled, not run on hardware
//
#include <hip/hip_runtime.h>
#include <cstdint>
#include <cstddef>

// ---------------------------------------------------------------------------
// Benes block for MI455X (gfx1250): 25 switch-unit levels, bf16 WMMA GEMMs
// with register-prefetch + double-buffered LDS pipelining.
// ---------------------------------------------------------------------------

typedef __bf16 bf16_t;
typedef __attribute__((ext_vector_type(16))) __bf16 bf16x16;
typedef __attribute__((ext_vector_type(8)))  float  f32x8;

#define LROWS   8192
#define NUF     512
#define MROWS   4096            // LROWS / 2
#define DIN     1024            // 2 * NUF
#define DHID    2048            // 4 * NUF
#define CANDW   0.10897247358851685f   // sqrt(1 - 0.9^2) * 0.25
#define LN_EPS  1e-6f
#define LDT     40              // padded LDS stride (bf16 units): conflict-free, 16B-aligned

// ---------------------------------------------------------------------------
// GEMM1: H[4096,2048] = A_f32[4096,1024] (cvt->bf16) @ W1_bf16[1024,2048]
// ---------------------------------------------------------------------------
__global__ __launch_bounds__(256)
void gemm1_kernel(const float* __restrict__ A,
                  const bf16_t* __restrict__ B,
                  float* __restrict__ H)
{
    __shared__ bf16_t As[2][128 * LDT];   // row-major [m][k]
    __shared__ bf16_t Bs[2][128 * LDT];   // transposed [n][k]

    const int t    = threadIdx.x;
    const int lane = t & 31;
    const int wave = t >> 5;
    const int M0   = blockIdx.y * 128;
    const int N0   = blockIdx.x * 128;

    // staging coordinates
    const int ra = t >> 3;          // 0..31  (A row, +32 per iter)
    const int ka = (t & 7) * 4;     // 0..28  (A k-col)
    const int kb = t >> 4;          // 0..15  (B k-row, +16 per iter)
    const int nb = (t & 15) * 8;    // 0..120 (B n-col)

    float4 pfA[4];
    uint4  pfB[2];

    auto ldg = [&](int k0) {
#pragma unroll
        for (int it = 0; it < 4; ++it)
            pfA[it] = *(const float4*)(&A[(size_t)(M0 + ra + it * 32) * DIN + k0 + ka]);
#pragma unroll
        for (int it = 0; it < 2; ++it)
            pfB[it] = *(const uint4*)(&B[(size_t)(k0 + kb + it * 16) * DHID + N0 + nb]);
    };
    auto sts = [&](int p) {
#pragma unroll
        for (int it = 0; it < 4; ++it) {
            bf16_t* d = &As[p][(ra + it * 32) * LDT + ka];
            d[0] = (bf16_t)pfA[it].x; d[1] = (bf16_t)pfA[it].y;
            d[2] = (bf16_t)pfA[it].z; d[3] = (bf16_t)pfA[it].w;
        }
#pragma unroll
        for (int it = 0; it < 2; ++it) {
            const bf16_t* e = reinterpret_cast<const bf16_t*>(&pfB[it]);
#pragma unroll
            for (int j = 0; j < 8; ++j)
                Bs[p][(nb + j) * LDT + kb + it * 16] = e[j];
        }
    };

    f32x8 acc[8];
#pragma unroll
    for (int i = 0; i < 8; ++i)
        acc[i] = (f32x8){0.f, 0.f, 0.f, 0.f, 0.f, 0.f, 0.f, 0.f};

    ldg(0);
    sts(0);
    __syncthreads();

    const int m    = wave * 16 + (lane & 15);
    const int ksel = lane >> 4;

    for (int k0 = 0, p = 0; k0 < DIN; k0 += 32, p ^= 1) {
        const bool more = (k0 + 32) < DIN;
        if (more) ldg(k0 + 32);               // prefetch next tile into registers

        bf16x16 afrag;
        {
            const bf16_t* pa = &As[p][m * LDT + ksel * 16];
#pragma unroll
            for (int i = 0; i < 16; ++i) afrag[i] = pa[i];
        }
#pragma unroll
        for (int nt = 0; nt < 8; ++nt) {
            bf16x16 bfrag;
            const bf16_t* pb = &Bs[p][(nt * 16 + (lane & 15)) * LDT + (lane >> 4) * 16];
#pragma unroll
            for (int i = 0; i < 16; ++i) bfrag[i] = pb[i];
            acc[nt] = __builtin_amdgcn_wmma_f32_16x16x32_bf16(
                false, afrag, false, bfrag, (short)0, acc[nt], false, false);
        }

        if (more) sts(p ^ 1);                 // fill the other buffer
        __syncthreads();                      // single barrier per K-step
    }

    // ---- writeback: C/D layout: VGPR i -> row i + 8*(lane>>4), col = lane&15
    const int mrow = M0 + wave * 16 + 8 * (lane >> 4);
    const int ncol = N0 + (lane & 15);
#pragma unroll
    for (int nt = 0; nt < 8; ++nt)
#pragma unroll
        for (int i = 0; i < 8; ++i)
            H[(size_t)(mrow + i) * DHID + ncol + nt * 16] = acc[nt][i];
}

// ---------------------------------------------------------------------------
// Column stats stage 1: per (64-col, 256-row) block partial sum / sumsq
// ---------------------------------------------------------------------------
__global__ __launch_bounds__(256)
void colstats_part_kernel(const float* __restrict__ H, float* __restrict__ part)
{
    __shared__ float ssum[4][64];
    __shared__ float ssq[4][64];
    const int cx = threadIdx.x & 63;
    const int ty = threadIdx.x >> 6;
    const int c  = blockIdx.x * 64 + cx;
    const int r0 = blockIdx.y * 256;
    float s = 0.f, q = 0.f;
    for (int r = r0 + ty; r < r0 + 256; r += 4) {
        float v = H[(size_t)r * DHID + c];
        s += v; q += v * v;
    }
    ssum[ty][cx] = s; ssq[ty][cx] = q;
    __syncthreads();
    if (ty == 0) {
        s = ssum[0][cx] + ssum[1][cx] + ssum[2][cx] + ssum[3][cx];
        q = ssq[0][cx] + ssq[1][cx] + ssq[2][cx] + ssq[3][cx];
        float* p = part + ((size_t)blockIdx.y * DHID + c) * 2;
        p[0] = s; p[1] = q;
    }
}

// ---------------------------------------------------------------------------
// Column stats stage 2: reduce 16 partials -> mean, rstd per column
// ---------------------------------------------------------------------------
__global__ __launch_bounds__(256)
void colstats_final_kernel(const float* __restrict__ part, float* __restrict__ stats)
{
    const int c = blockIdx.x * 256 + threadIdx.x;
    if (c >= DHID) return;
    float s = 0.f, q = 0.f;
#pragma unroll
    for (int b = 0; b < 16; ++b) {
        const float* p = part + ((size_t)b * DHID + c) * 2;
        s += p[0]; q += p[1];
    }
    const float mean = s * (1.0f / MROWS);
    const float var  = q * (1.0f / MROWS) - mean * mean;
    stats[2 * c]     = mean;
    stats[2 * c + 1] = rsqrtf(var + LN_EPS);
}

// ---------------------------------------------------------------------------
// Normalize + leaky-ReLU(0.2) + convert to bf16
// ---------------------------------------------------------------------------
__global__ __launch_bounds__(256)
void norm_kernel(const float* __restrict__ H, const float* __restrict__ stats,
                 bf16_t* __restrict__ HB)
{
    const size_t g = ((size_t)blockIdx.x * 256 + threadIdx.x) * 4;
    const int c0 = (int)(g & (DHID - 1));
    const float4 v = *(const float4*)(&H[g]);
    float e[4] = {v.x, v.y, v.z, v.w};
#pragma unroll
    for (int j = 0; j < 4; ++j) {
        const float mean = stats[2 * (c0 + j)];
        const float rstd = stats[2 * (c0 + j) + 1];
        float x = (e[j] - mean) * rstd;
        x = (x > 0.f) ? x : 0.2f * x;
        HB[g + j] = (bf16_t)x;
    }
}

// ---------------------------------------------------------------------------
// GEMM2: cand = HB[4096,2048] @ W2_bf16[2048,1024]; fused epilogue:
//   out[perm(row), col] = sig(rs[col]) * xin[row,col] + (cand + b2) * CANDW
// mode: 0 = scatter ROL (forward), 1 = scatter ROR (reverse), 2 = identity
// ---------------------------------------------------------------------------
__global__ __launch_bounds__(256)
void gemm2_kernel(const bf16_t* __restrict__ A,
                  const bf16_t* __restrict__ B,
                  const float* __restrict__ b2,
                  const float* __restrict__ sigs,
                  const float* __restrict__ xin,
                  float* __restrict__ out,
                  int mode)
{
    __shared__ bf16_t As[2][128 * LDT];
    __shared__ bf16_t Bs[2][128 * LDT];

    const int t    = threadIdx.x;
    const int lane = t & 31;
    const int wave = t >> 5;
    const int M0   = blockIdx.y * 128;
    const int N0   = blockIdx.x * 128;

    const int ra = t >> 2;          // 0..63  (A row, +64 per iter)
    const int ka = (t & 3) * 8;     // 0,8,16,24
    const int kb = t >> 4;          // 0..15
    const int nb = (t & 15) * 8;    // 0..120

    uint4 pfA[2];
    uint4 pfB[2];

    auto ldg = [&](int k0) {
#pragma unroll
        for (int it = 0; it < 2; ++it)
            pfA[it] = *(const uint4*)(&A[(size_t)(M0 + ra + it * 64) * DHID + k0 + ka]);
#pragma unroll
        for (int it = 0; it < 2; ++it)
            pfB[it] = *(const uint4*)(&B[(size_t)(k0 + kb + it * 16) * DIN + N0 + nb]);
    };
    auto sts = [&](int p) {
#pragma unroll
        for (int it = 0; it < 2; ++it)
            *(uint4*)(&As[p][(ra + it * 64) * LDT + ka]) = pfA[it];
#pragma unroll
        for (int it = 0; it < 2; ++it) {
            const bf16_t* e = reinterpret_cast<const bf16_t*>(&pfB[it]);
#pragma unroll
            for (int j = 0; j < 8; ++j)
                Bs[p][(nb + j) * LDT + kb + it * 16] = e[j];
        }
    };

    f32x8 acc[8];
#pragma unroll
    for (int i = 0; i < 8; ++i)
        acc[i] = (f32x8){0.f, 0.f, 0.f, 0.f, 0.f, 0.f, 0.f, 0.f};

    ldg(0);
    sts(0);
    __syncthreads();

    const int m    = wave * 16 + (lane & 15);
    const int ksel = lane >> 4;

    for (int k0 = 0, p = 0; k0 < DHID; k0 += 32, p ^= 1) {
        const bool more = (k0 + 32) < DHID;
        if (more) ldg(k0 + 32);

        bf16x16 afrag;
        {
            const bf16_t* pa = &As[p][m * LDT + ksel * 16];
#pragma unroll
            for (int i = 0; i < 16; ++i) afrag[i] = pa[i];
        }
#pragma unroll
        for (int nt = 0; nt < 8; ++nt) {
            bf16x16 bfrag;
            const bf16_t* pb = &Bs[p][(nt * 16 + (lane & 15)) * LDT + (lane >> 4) * 16];
#pragma unroll
            for (int i = 0; i < 16; ++i) bfrag[i] = pb[i];
            acc[nt] = __builtin_amdgcn_wmma_f32_16x16x32_bf16(
                false, afrag, false, bfrag, (short)0, acc[nt], false, false);
        }

        if (more) sts(p ^ 1);
        __syncthreads();
    }

    // ---- fused epilogue: bias, residual, candidate weight, row scatter
    const int p_base = M0 + wave * 16 + 8 * (lane >> 4);   // row in [0,4096)
#pragma unroll
    for (int nt = 0; nt < 8; ++nt) {
        const int c = N0 + nt * 16 + (lane & 15);          // col in [0,1024)
        const int chalf = c >> 9;                          // 0 or 1 (tile never straddles)
        const int col   = c & (NUF - 1);
        const float bias = b2[c];
        const float sg   = sigs[col];
#pragma unroll
        for (int i = 0; i < 8; ++i) {
            const int pr  = p_base + i;
            const int row = 2 * pr + chalf;                // row in [0,8192)
            const float val = sg * xin[(size_t)row * NUF + col]
                            + (acc[nt][i] + bias) * CANDW;
            int dest;
            if (mode == 0)      dest = (row < MROWS) ? 2 * row : 2 * row - (LROWS - 1); // ROL
            else if (mode == 1) dest = (row >> 1) + ((row & 1) ? MROWS : 0);            // ROR
            else                dest = row;
            out[(size_t)dest * NUF + col] = val;
        }
    }
}

// ---------------------------------------------------------------------------
// Helpers: f32 -> bf16 weight conversion, sigmoid(rs) precompute
// ---------------------------------------------------------------------------
__global__ __launch_bounds__(256)
void f32_to_bf16_kernel(const float* __restrict__ src, bf16_t* __restrict__ dst, int n)
{
    const int i = (blockIdx.x * 256 + threadIdx.x) * 4;
    if (i + 3 < n) {
        const float4 v = *(const float4*)(src + i);
        dst[i]     = (bf16_t)v.x;
        dst[i + 1] = (bf16_t)v.y;
        dst[i + 2] = (bf16_t)v.z;
        dst[i + 3] = (bf16_t)v.w;
    } else {
        for (int j = i; j < n; ++j) dst[j] = (bf16_t)src[j];
    }
}

__global__ __launch_bounds__(256)
void sigmoid_kernel(const float* __restrict__ rs, float* __restrict__ sigs, int n)
{
    const int i = blockIdx.x * 256 + threadIdx.x;
    if (i < n) sigs[i] = 1.0f / (1.0f + __expf(-rs[i]));
}

// ---------------------------------------------------------------------------
// Host-side orchestration
// ---------------------------------------------------------------------------
extern "C" void kernel_launch(void* const* d_in, const int* in_sizes, int n_in,
                              void* d_out, int out_size, void* d_ws, size_t ws_size,
                              hipStream_t stream)
{
    const float* x = (const float*)d_in[0];
    const float* rs[3] = {(const float*)d_in[1], (const float*)d_in[5], (const float*)d_in[9]};
    const float* w1[3] = {(const float*)d_in[2], (const float*)d_in[6], (const float*)d_in[10]};
    const float* w2[3] = {(const float*)d_in[3], (const float*)d_in[7], (const float*)d_in[11]};
    const float* b2[3] = {(const float*)d_in[4], (const float*)d_in[8], (const float*)d_in[12]};

    char* ws = (char*)d_ws;
    size_t off = 0;
    float*  yA    = (float*)(ws + off);  off += (size_t)LROWS * NUF * 4;       // 16 MB
    float*  yB    = (float*)(ws + off);  off += (size_t)LROWS * NUF * 4;       // 16 MB
    float*  H     = (float*)(ws + off);  off += (size_t)MROWS * DHID * 4;      // 32 MB
    bf16_t* HB    = (bf16_t*)(ws + off); off += (size_t)MROWS * DHID * 2;      // 16 MB
    bf16_t* w1b[3];
    for (int s = 0; s < 3; ++s) { w1b[s] = (bf16_t*)(ws + off); off += (size_t)DIN * DHID * 2; }
    bf16_t* w2b[3];
    for (int s = 0; s < 3; ++s) { w2b[s] = (bf16_t*)(ws + off); off += (size_t)DHID * DIN * 2; }
    float*  stats = (float*)(ws + off);  off += (size_t)DHID * 2 * 4;
    float*  part  = (float*)(ws + off);  off += (size_t)16 * DHID * 2 * 4;
    float*  sigs  = (float*)(ws + off);  off += (size_t)3 * NUF * 4;

    // One-time per call: bf16 weights + sigmoid(rs)
    const int nw = DIN * DHID;                 // 2,097,152
    for (int s = 0; s < 3; ++s) {
        f32_to_bf16_kernel<<<nw / 1024, 256, 0, stream>>>(w1[s], w1b[s], nw);
        f32_to_bf16_kernel<<<nw / 1024, 256, 0, stream>>>(w2[s], w2b[s], nw);
        sigmoid_kernel<<<2, 256, 0, stream>>>(rs[s], sigs + (size_t)s * NUF, NUF);
    }

    const dim3 g1(DHID / 128, MROWS / 128);    // (16, 32)
    const dim3 g2(DIN / 128, MROWS / 128);     // (8, 32)
    const dim3 gs(DHID / 64, 16);              // (32, 16)

    const float* cur = x;
    for (int lev = 0; lev < 25; ++lev) {
        const int s    = (lev < 12) ? 0 : ((lev < 24) ? 1 : 2);
        const int mode = (lev < 12) ? 0 : ((lev < 24) ? 1 : 2);
        float* outp = (lev == 24) ? (float*)d_out : ((lev & 1) ? yB : yA);

        gemm1_kernel<<<g1, 256, 0, stream>>>(cur, w1b[s], H);
        colstats_part_kernel<<<gs, 256, 0, stream>>>(H, part);
        colstats_final_kernel<<<DHID / 256, 256, 0, stream>>>(part, stats);
        norm_kernel<<<(MROWS * DHID) / 1024, 256, 0, stream>>>(H, stats, HB);
        gemm2_kernel<<<g2, 256, 0, stream>>>(HB, w2b[s], b2[s],
                                             sigs + (size_t)s * NUF, cur, outp, mode);
        cur = outp;
    }
}
